// QualityRegularizationLoss_12189117186266
// MI455X (gfx1250) — matrix operand order repaired and checked by reading, not compiled, not run on hardware
//
#include <hip/hip_runtime.h>
#include <hip/hip_bf16.h>

// Problem constants (match reference)
#define BATCH   16384
#define DIMK    1024
#define NTYPES  128      // softmax width T
#define NIDX    12       // V*L indices per sample
#define KCHUNK  32
#define NCHUNK  (DIMK / KCHUNK)              // 32
#define CHUNK_BYTES (NTYPES * KCHUNK * 2)    // 8192 B of bf16 per K-chunk

// Block shape: 64 rows, 4 wave32s (16 rows per wave) -> 256 workgroups,
// enough to fill all WGPs with multiple blocks each for latency hiding.
#define ROWS_BLK 64
#define NWAVES   4
#define NTHREADS 128

// CDNA5 WMMA vector types
typedef __attribute__((ext_vector_type(16))) __bf16 v16bf;
typedef __attribute__((ext_vector_type(2)))  __bf16 v2bf;
typedef __attribute__((ext_vector_type(8)))  float  v8f;

union BF16Frag {
    v16bf v;
    unsigned int u[8];
    uint4 q[2];
};

// packed fp32x2 -> bf16x2 (RNE) using hardware convert when available
__device__ __forceinline__ unsigned int pk2(float lo, float hi) {
#if __has_builtin(__builtin_amdgcn_cvt_pk_bf16_f32)
    v2bf r = __builtin_amdgcn_cvt_pk_bf16_f32(lo, hi);
    unsigned int u;
    __builtin_memcpy(&u, &r, 4);
    return u;
#else
    __hip_bfloat162 h = __float22bfloat162_rn(make_float2(lo, hi));
    unsigned int u;
    __builtin_memcpy(&u, &h, 4);
    return u;
#endif
}

// ---- async global->LDS copy (16 B per lane), with synchronous fallback ----
#if __has_builtin(__builtin_amdgcn_global_load_async_to_lds_b128)
#define HAS_ASYNC_LDS 1
#else
#define HAS_ASYNC_LDS 0
#endif

// builtin parameter type: int vector_size(16), AS1 source / AS3 dest
typedef int v4i_a __attribute__((__vector_size__(16)));

__device__ __forceinline__ void copy16_g2l(const void* g, void* l) {
#if HAS_ASYNC_LDS
    __builtin_amdgcn_global_load_async_to_lds_b128(
        (__attribute__((address_space(1))) v4i_a*)g,
        (__attribute__((address_space(3))) v4i_a*)l, 0, 0);
#else
    *(uint4*)l = *(const uint4*)g;
#endif
}
__device__ __forceinline__ void async_fence() {
#if HAS_ASYNC_LDS && __has_builtin(__builtin_amdgcn_s_wait_asynccnt)
    __builtin_amdgcn_s_wait_asynccnt(0);
#endif
}

// LDS layout (dynamic shared):
//   bf16 W double buffer: 128 rows (type t), 32 bf16 data + pad, row stride
//   112 B (28 dwords; gcd(28,64)=4 -> 16 consecutive rows -> 16 distinct
//   banks). 2 x 128 x 112 B = 28672 B.
//   After the K loop the region is reused for probs: 64 rows x 132 f32.
#define WROW_BYTES   112
#define WBUF_BYTES   (128 * WROW_BYTES)              // 14336
#define PROB_STRIDE  132
#define PROB_BYTES   (ROWS_BLK * PROB_STRIDE * 4)    // 33792
#define RED_OFF      PROB_BYTES
#define SMEM_BYTES   (PROB_BYTES + 64)

// ---------------------------------------------------------------------------
// Kernel 0: convert W [T, D] fp32 -> chunk-major bf16: [chunk][t][32 k]
// ---------------------------------------------------------------------------
__global__ void __launch_bounds__(256)
qrl_pack_w(const float* __restrict__ W, unsigned int* __restrict__ wbf)
{
    const int chunk = blockIdx.x;           // 0..31
    const int wt = threadIdx.x >> 1;        // 0..127
    const int wk = (threadIdx.x & 1) * 16;  // 0 or 16
    const float* src = W + wt * DIMK + chunk * KCHUNK + wk;
    float4 f0 = ((const float4*)src)[0];
    float4 f1 = ((const float4*)src)[1];
    float4 f2 = ((const float4*)src)[2];
    float4 f3 = ((const float4*)src)[3];
    uint4* dst = (uint4*)(wbf + chunk * (NTYPES * KCHUNK / 2) + wt * (KCHUNK / 2)
                          + (wk >> 1));
    dst[0] = make_uint4(pk2(f0.x, f0.y), pk2(f0.z, f0.w),
                        pk2(f1.x, f1.y), pk2(f1.z, f1.w));
    dst[1] = make_uint4(pk2(f2.x, f2.y), pk2(f2.z, f2.w),
                        pk2(f3.x, f3.y), pk2(f3.z, f3.w));
}

// ---------------------------------------------------------------------------
// Kernel 1: fused GEMM (bf16 WMMA, f32 accum) + softmax + gather + partials
// ---------------------------------------------------------------------------
__global__ void __launch_bounds__(NTHREADS)
qrl_fused_kernel(const float*        __restrict__ A,    // [B, D] fp32
                 const unsigned int* __restrict__ wbf,  // packed bf16 W chunks
                 const float*        __restrict__ bias, // [T]
                 const int*          __restrict__ idx,  // [B, 12]
                 const int*          __restrict__ msk,  // [B, 12]
                 float*              __restrict__ partials)
{
    extern __shared__ unsigned char smem[];
    float* redbuf = (float*)(smem + RED_OFF);

    const int tid  = threadIdx.x;
    const int lane = tid & 31;
    const int wave = tid >> 5;       // 4 waves
    const int l16  = lane & 15;
    const int half = lane >> 4;      // wave32 halves

    const int rowBlock = blockIdx.x * ROWS_BLK;
    const int rowWave  = rowBlock + wave * 16;

    v8f c[8] = {};   // 8 N-tiles of 16x16 f32 accumulators

    // staging map: 512 x 16 B segments per chunk, 4 per thread
    const int s0 = tid,       s1 = tid + 128, s2 = tid + 256, s3 = tid + 384;
    const int t0 = s0 >> 2, p0 = s0 & 3;
    const int t1 = s1 >> 2, p1 = s1 & 3;
    const int t2 = s2 >> 2, p2 = s2 & 3;
    const int t3 = s3 >> 2, p3 = s3 & 3;

    auto stage = [&](int chunk, int buf) {
        const unsigned char* g = (const unsigned char*)wbf + chunk * CHUNK_BYTES;
        unsigned char* l = smem + buf * WBUF_BYTES;
        copy16_g2l(g + t0 * 64 + p0 * 16, l + t0 * WROW_BYTES + p0 * 16);
        copy16_g2l(g + t1 * 64 + p1 * 16, l + t1 * WROW_BYTES + p1 * 16);
        copy16_g2l(g + t2 * 64 + p2 * 16, l + t2 * WROW_BYTES + p2 * 16);
        copy16_g2l(g + t3 * 64 + p3 * 16, l + t3 * WROW_BYTES + p3 * 16);
    };

    auto loadfrag = [&](BF16Frag& b, const unsigned char* wbase, int nt) {
        const uint4* src = (const uint4*)(wbase + (nt * 16 + l16) * WROW_BYTES
                                          + half * 16);
        b.q[0] = src[0];
        b.q[1] = src[2];
    };

    stage(0, 0);

    // A-row stream, software pipelined (raw fp32, convert in-register)
    const float* aRow = A + (long)(rowWave + l16) * DIMK;
    float4 an0, an1, an2, an3;
    {
        const float* ap = aRow + half * 8;
        an0 = ((const float4*)ap)[0];
        an1 = ((const float4*)ap)[1];
        an2 = ((const float4*)(ap + 16))[0];
        an3 = ((const float4*)(ap + 16))[1];
    }

    async_fence();
    __syncthreads();

    for (int kc = 0; kc < NCHUNK; ++kc) {
        const int  cur  = kc & 1;
        const bool more = (kc + 1) < NCHUNK;

        if (more) stage(kc + 1, cur ^ 1);  // async copy next W chunk

        // A fragment: convert straight out of the prefetch registers
        // (16x32 bf16: lane holds row M=l16, K = kb..kb+7, kb+16..kb+23,
        //  kb = half*8)
        BF16Frag af;
        af.u[0] = pk2(an0.x, an0.y); af.u[1] = pk2(an0.z, an0.w);
        af.u[2] = pk2(an1.x, an1.y); af.u[3] = pk2(an1.z, an1.w);
        af.u[4] = pk2(an2.x, an2.y); af.u[5] = pk2(an2.z, an2.w);
        af.u[6] = pk2(an3.x, an3.y); af.u[7] = pk2(an3.z, an3.w);

        // prefetch registers are free again: issue next A loads
        if (more) {
            const float* ap = aRow + (kc + 1) * KCHUNK + half * 8;
            an0 = ((const float4*)ap)[0];
            an1 = ((const float4*)ap)[1];
            an2 = ((const float4*)(ap + 16))[0];
            an3 = ((const float4*)(ap + 16))[1];
        }

        // B-fragment two-buffer pipeline + 8 WMMAs
        const unsigned char* wbase = smem + cur * WBUF_BYTES;
        BF16Frag b0, b1;
        loadfrag(b0, wbase, 0);
        #pragma unroll
        for (int nt = 0; nt < 8; nt += 2) {
            loadfrag(b1, wbase, nt + 1);
            c[nt] = __builtin_amdgcn_wmma_f32_16x16x32_bf16(
                        false, af.v, false, b0.v, (short)0, c[nt],
                        false, false);
            if (nt + 2 < 8) loadfrag(b0, wbase, nt + 2);
            c[nt + 1] = __builtin_amdgcn_wmma_f32_16x16x32_bf16(
                        false, af.v, false, b1.v, (short)0, c[nt + 1],
                        false, false);
        }

        if (more) async_fence();
        __syncthreads();
    }

    // ---- bias + softmax over T=128, per register-row r ----
    float bb[8];
    #pragma unroll
    for (int nt = 0; nt < 8; ++nt) bb[nt] = bias[nt * 16 + l16];
    #pragma unroll
    for (int nt = 0; nt < 8; ++nt)
        #pragma unroll
        for (int r = 0; r < 8; ++r) c[nt][r] += bb[nt];

    float rinv[8];
    #pragma unroll
    for (int r = 0; r < 8; ++r) {
        float m = c[0][r];
        #pragma unroll
        for (int nt = 1; nt < 8; ++nt) m = fmaxf(m, c[nt][r]);
        #pragma unroll
        for (int o = 1; o < 16; o <<= 1) m = fmaxf(m, __shfl_xor(m, o, 32));
        float s = 0.0f;
        #pragma unroll
        for (int nt = 0; nt < 8; ++nt) {
            float e = __expf(c[nt][r] - m);
            c[nt][r] = e;
            s += e;
        }
        #pragma unroll
        for (int o = 1; o < 16; o <<= 1) s += __shfl_xor(s, o, 32);
        rinv[r] = 1.0f / s;
    }

    // ---- spill normalized probs to LDS (W buffers are dead now) ----
    float* probs = (float*)smem;
    #pragma unroll
    for (int nt = 0; nt < 8; ++nt)
        #pragma unroll
        for (int r = 0; r < 8; ++r) {
            const int lr = wave * 16 + half * 8 + r;   // local row 0..63
            probs[lr * PROB_STRIDE + nt * 16 + l16] = c[nt][r] * rinv[r];
        }
    __syncthreads();

    // ---- gather: one thread per row ----
    float acc = 0.0f;
    if (tid < ROWS_BLK) {
        const int row = rowBlock + tid;
        const int* ip = idx + row * NIDX;
        const int* mp = msk + row * NIDX;
        const float* pr = probs + tid * PROB_STRIDE;
        #pragma unroll
        for (int j = 0; j < NIDX; ++j)
            acc += mp[j] ? pr[ip[j]] : 0.0f;
    }

    // ---- block reduce ----
    #pragma unroll
    for (int o = 16; o >= 1; o >>= 1) acc += __shfl_xor(acc, o, 32);
    if (lane == 0) redbuf[wave] = acc;
    __syncthreads();
    if (tid == 0) {
        float t = 0.0f;
        #pragma unroll
        for (int w = 0; w < NWAVES; ++w) t += redbuf[w];
        partials[blockIdx.x] = t;
    }
}

// ---------------------------------------------------------------------------
// Kernel 2: deterministic final reduction of 256 partials
// ---------------------------------------------------------------------------
__global__ void qrl_reduce_kernel(const float* __restrict__ partials,
                                  float* __restrict__ out)
{
    const int l = threadIdx.x;  // 32 threads
    float v = 0.0f;
    #pragma unroll
    for (int i = 0; i < 8; ++i) v += partials[l + 32 * i];
    #pragma unroll
    for (int o = 16; o >= 1; o >>= 1) v += __shfl_xor(v, o, 32);
    if (l == 0) out[0] = v * (1.0f / (4.0f * (float)BATCH));  // /V /B
}

extern "C" void kernel_launch(void* const* d_in, const int* in_sizes, int n_in,
                              void* d_out, int out_size, void* d_ws, size_t ws_size,
                              hipStream_t stream) {
    (void)in_sizes; (void)n_in; (void)out_size; (void)ws_size;
    const float* A    = (const float*)d_in[0];   // clean_embeddings [B, D]
    const float* W    = (const float*)d_in[1];   // [T, D]
    const float* bias = (const float*)d_in[2];   // [T]
    const int*   idx  = (const int*)d_in[3];     // [B, V, L]
    const int*   msk  = (const int*)d_in[4];     // [B, V, L]
    float* out = (float*)d_out;

    // workspace: [0, 1024)  per-block partials (256 floats)
    //            [4096, +256KB) packed bf16 W (needs ws_size >= ~261 KB)
    float*        partials = (float*)d_ws;
    unsigned int* wbf      = (unsigned int*)((char*)d_ws + 4096);

    qrl_pack_w<<<dim3(NCHUNK), dim3(256), 0, stream>>>(W, wbf);
    qrl_fused_kernel<<<dim3(BATCH / ROWS_BLK), dim3(NTHREADS), SMEM_BYTES,
                       stream>>>(A, wbf, bias, idx, msk, partials);
    qrl_reduce_kernel<<<dim3(1), dim3(32), 0, stream>>>(partials, out);
}
